// DeepSeekV31DecoderBlock_82145544504075
// MI455X (gfx1250) — matrix operand order repaired and checked by reading, not compiled, not run on hardware
//
#include <hip/hip_runtime.h>
#include <hip/hip_bf16.h>

// ---------------------------------------------------------------------------
// DeepSeek-V3 MLA decoder block for gfx1250 (MI455X).
// - Weights pre-converted to bf16 and pre-transposed to [N][K] once per call.
// - All GEMMs: bf16 operands via GLOBAL_LOAD_ASYNC_TO_LDS_B128 staging,
//   V_WMMA_F32_16X16X32_BF16 compute, f32 accumulate/output.
// - Flash attention with the same async staging + WMMA.
// Requires ws_size >= ~460 MB.
// ---------------------------------------------------------------------------

typedef __attribute__((ext_vector_type(16))) __bf16 bf16x16;
typedef __attribute__((ext_vector_type(8)))  float  f32x8;

__device__ __forceinline__ unsigned short f2bf(float f) {
  unsigned u = __float_as_uint(f);
  unsigned r = (u >> 16) & 1u;
  u += 0x7fffu + r;
  return (unsigned short)(u >> 16);
}

// Pack two floats into a bf16x2 dword (v_cvt_pk_bf16_f32, assembler-verified).
__device__ __forceinline__ unsigned pack2(float lo, float hi) {
#if defined(__gfx1250__)
  unsigned r;
  asm("v_cvt_pk_bf16_f32 %0, %1, %2" : "=v"(r) : "v"(lo), "v"(hi));
  return r;
#else
  return (unsigned)f2bf(lo) | ((unsigned)f2bf(hi) << 16);
#endif
}

__device__ __forceinline__ f32x8 zero8() {
  f32x8 z;
#pragma unroll
  for (int i = 0; i < 8; i++) z[i] = 0.0f;
  return z;
}

// LDS byte offset of a generic pointer to __shared__ (its low 32 bits).
__device__ __forceinline__ unsigned lds_off(const void* p) {
  return (unsigned)(unsigned long long)p;
}

// Async copy: 16 bytes global -> LDS per lane, tracked by ASYNCcnt.
__device__ __forceinline__ void async_copy_b128(unsigned lds_byte_off, const void* gsrc) {
  asm volatile("global_load_async_to_lds_b128 %0, %1, off"
               :: "v"(lds_byte_off), "v"((unsigned long long)gsrc)
               : "memory");
}
__device__ __forceinline__ void wait_async0() {
  asm volatile("s_wait_asynccnt 0x0" ::: "memory");
}

union FragU { uint4 u[2]; bf16x16 v; };

// A fragment (16x32 bf16, M across lanes 0..15, K split by half-wave).
// LDS layout: row-major [row][k], row stride = ldk (multiple of 8).
__device__ __forceinline__ bf16x16 load_frag_a(const unsigned short* base,
                                               int ldk, int row0, int kofs) {
  int l = threadIdx.x & 31;
  int m = l & 15, g = l >> 4;
  const unsigned short* p = base + (row0 + m) * ldk + kofs;
  FragU f;
  f.u[0] = *(const uint4*)(p + g * 8);        // K = g*8 + 0..7
  f.u[1] = *(const uint4*)(p + 16 + g * 8);   // K = 16 + g*8 + 0..7
  return f.v;
}

// B fragment (32x16 bf16, N across lanes 0..15, K split by half-wave).
// LDS layout: [n][k] (B transposed), row stride = ldk (multiple of 8).
__device__ __forceinline__ bf16x16 load_frag_b(const unsigned short* base,
                                               int ldk, int col0, int kofs) {
  int l = threadIdx.x & 31;
  int n = l & 15, g = l >> 4;
  const unsigned short* p = base + (col0 + n) * ldk + kofs;
  FragU f;
  f.u[0] = *(const uint4*)(p + g * 8);        // K = g*8 + 0..7
  f.u[1] = *(const uint4*)(p + 16 + g * 8);   // K = 16 + g*8 + 0..7
  return f.v;
}

// ---------------------------------------------------------------------------
// Weight convert+transpose: W f32 [K][N] -> Wt bf16 [Npad][K] (zero pad rows).
// Grid: (K/64, Npad/64); K, N multiples of 64.
// ---------------------------------------------------------------------------
__global__ __launch_bounds__(256) void transpose_bf16_kernel(
    const float* __restrict__ W, unsigned short* __restrict__ Wt,
    int K, int N, int Npad) {
  __shared__ __align__(16) unsigned short T[64 * 64];   // [kk][nn]
  int k0 = blockIdx.x * 64, n1 = blockIdx.y * 64;
  int t = threadIdx.x;
  bool valid = n1 < N;
  if (valid) {
    int kk = t >> 2, nn = (t & 3) * 16;
    const float4* wp = (const float4*)(W + (size_t)(k0 + kk) * N + n1 + nn);
    float4 f0 = wp[0], f1 = wp[1], f2v = wp[2], f3 = wp[3];
    uint4 p0, p1;
    p0.x = pack2(f0.x, f0.y);   p0.y = pack2(f0.z, f0.w);
    p0.z = pack2(f1.x, f1.y);   p0.w = pack2(f1.z, f1.w);
    p1.x = pack2(f2v.x, f2v.y); p1.y = pack2(f2v.z, f2v.w);
    p1.z = pack2(f3.x, f3.y);   p1.w = pack2(f3.z, f3.w);
    *(uint4*)(T + kk * 64 + nn) = p0;
    *(uint4*)(T + kk * 64 + nn + 8) = p1;
  }
  __syncthreads();
  int nn2 = t >> 2, kb = (t & 3) * 16;
  unsigned outp[8];
  if (valid) {
#pragma unroll
    for (int i = 0; i < 8; i++) {
      unsigned short lo = T[(kb + 2 * i) * 64 + nn2];
      unsigned short hi = T[(kb + 2 * i + 1) * 64 + nn2];
      outp[i] = (unsigned)lo | ((unsigned)hi << 16);
    }
  } else {
#pragma unroll
    for (int i = 0; i < 8; i++) outp[i] = 0u;
  }
  unsigned short* dp = Wt + (size_t)(n1 + nn2) * K + k0 + kb;
  *(uint4*)(dp) = *(uint4*)(outp);
  *(uint4*)(dp + 8) = *(uint4*)(outp + 4);
}

// ---------------------------------------------------------------------------
// RMSNorm producing bf16: one block per row.
// ---------------------------------------------------------------------------
__global__ __launch_bounds__(256) void rmsnorm_bf16_kernel(
    const float* __restrict__ in, unsigned short* __restrict__ out,
    const float* __restrict__ scale, int cols, int in_stride, int out_stride) {
  int row = blockIdx.x;
  const float* x = in + (size_t)row * in_stride;
  unsigned short* y = out + (size_t)row * out_stride;
  float ss = 0.0f;
  for (int c = threadIdx.x; c < cols; c += 256) { float v = x[c]; ss += v * v; }
  __shared__ float red[256];
  red[threadIdx.x] = ss;
  __syncthreads();
  for (int s = 128; s > 0; s >>= 1) {
    if (threadIdx.x < s) red[threadIdx.x] += red[threadIdx.x + s];
    __syncthreads();
  }
  float r = rsqrtf(red[0] / (float)cols + 1e-6f);
  for (int c = threadIdx.x * 2; c < cols; c += 512) {
    float v0 = x[c] * r * scale[c];
    float v1 = x[c + 1] * r * scale[c + 1];
    *(unsigned*)(y + c) = pack2(v0, v1);
  }
}

// ---------------------------------------------------------------------------
// GEMM: C[M,N] = (Res ? Res : 0) + A[M,K](bf16,lda) @ Bt[N,K](bf16)^T.
// 256 threads (8 waves, 2x4 wave grid), 128x128x32 tiles, double-buffered LDS,
// both tiles staged with async b128 copies. M%128==0, K%32==0, Bt rows padded
// to >= ceil(N/128)*128 (pad rows zero). C stores guarded by N.
// ---------------------------------------------------------------------------
#define GT_BM 128
#define GT_BN 128
#define GT_BK 32

__device__ __forceinline__ void gemm_stage_async(
    const unsigned short* __restrict__ A, const unsigned short* __restrict__ Bt,
    int lda, int K, int m0, int n0, int k0,
    unsigned asbase, unsigned bsbase, int t) {
#pragma unroll
  for (int i = 0; i < 2; i++) {
    int u = t + 256 * i;                 // 512 uint4s per tile
    int row = u >> 2, ch = (u & 3) * 8;  // 4 x 8-elem chunks per 32-elem row
    async_copy_b128(asbase + (unsigned)(row * GT_BK + ch) * 2,
                    A + (size_t)(m0 + row) * lda + k0 + ch);
    async_copy_b128(bsbase + (unsigned)(row * GT_BK + ch) * 2,
                    Bt + (size_t)(n0 + row) * K + k0 + ch);
  }
}

__global__ __launch_bounds__(256) void gemm_bf16_kernel(
    const unsigned short* __restrict__ A, const unsigned short* __restrict__ Bt,
    const float* __restrict__ Res, float* __restrict__ C,
    int M, int N, int K, int lda) {
  __shared__ __align__(16) unsigned short As[2][GT_BM * GT_BK];
  __shared__ __align__(16) unsigned short Bs[2][GT_BN * GT_BK];
  int t = threadIdx.x;
  int m0 = blockIdx.x * GT_BM;
  int n0 = blockIdx.y * GT_BN;
  int w = t >> 5;
  int wm = w >> 2, wn = w & 3;   // 2 x 4 wave grid -> 64x32 per wave

  f32x8 acc[4][2];
#pragma unroll
  for (int i = 0; i < 4; i++)
#pragma unroll
    for (int j = 0; j < 2; j++) acc[i][j] = zero8();

  gemm_stage_async(A, Bt, lda, K, m0, n0, 0, lds_off(As[0]), lds_off(Bs[0]), t);
  wait_async0();
  __syncthreads();

  int nk = K / GT_BK;
  for (int ki = 0; ki < nk; ki++) {
    int cur = ki & 1;
    bool more = (ki + 1) < nk;
    if (more)  // async copies land in the other buffer while we compute
      gemm_stage_async(A, Bt, lda, K, m0, n0, (ki + 1) * GT_BK,
                       lds_off(As[cur ^ 1]), lds_off(Bs[cur ^ 1]), t);

    const unsigned short* Ac = As[cur];
    const unsigned short* Bc = Bs[cur];
    bf16x16 afr[4];
#pragma unroll
    for (int i = 0; i < 4; i++)
      afr[i] = load_frag_a(Ac, GT_BK, wm * 64 + i * 16, 0);
    bf16x16 bfr[2];
#pragma unroll
    for (int j = 0; j < 2; j++)
      bfr[j] = load_frag_b(Bc, GT_BK, wn * 32 + j * 16, 0);
#pragma unroll
    for (int i = 0; i < 4; i++)
#pragma unroll
      for (int j = 0; j < 2; j++)
        acc[i][j] = __builtin_amdgcn_wmma_f32_16x16x32_bf16(
            false, afr[i], false, bfr[j], (short)0, acc[i][j], false, false);

    if (more) wait_async0();
    __syncthreads();
  }

  // Epilogue: branch on Res once (uniform), straight-line store loops.
  int l = t & 31;
  int lm = l & 15, g = l >> 4;
  if (Res != nullptr) {
#pragma unroll
    for (int i = 0; i < 4; i++)
#pragma unroll
      for (int j = 0; j < 2; j++) {
        int col = n0 + wn * 32 + j * 16 + lm;
        if (col < N) {
          size_t rowbase = (size_t)(m0 + wm * 64 + i * 16 + g * 8);
#pragma unroll
          for (int r = 0; r < 8; r++) {
            size_t idx = (rowbase + r) * (size_t)N + col;
            C[idx] = acc[i][j][r] + Res[idx];
          }
        }
      }
  } else {
#pragma unroll
    for (int i = 0; i < 4; i++)
#pragma unroll
      for (int j = 0; j < 2; j++) {
        int col = n0 + wn * 32 + j * 16 + lm;
        if (col < N) {
          size_t rowbase = (size_t)(m0 + wm * 64 + i * 16 + g * 8);
#pragma unroll
          for (int r = 0; r < 8; r++) {
            size_t idx = (rowbase + r) * (size_t)N + col;
            C[idx] = acc[i][j][r];
          }
        }
      }
  }
}

// ---------------------------------------------------------------------------
// Build bf16 Q/K/V head-major tensors with RoPE applied.
// Grid: B*S*H blocks of 192 threads. Q/K [b][h][s][192], V [b][h][s][128].
// ---------------------------------------------------------------------------
__global__ __launch_bounds__(192) void build_qkv_kernel(
    const float* __restrict__ q, const float* __restrict__ kv,
    const float* __restrict__ kva, unsigned short* __restrict__ Qb,
    unsigned short* __restrict__ Kb, unsigned short* __restrict__ Vb) {
  const int Sdim = 2048;
  int bid = blockIdx.x;
  int h = bid & 15;
  int bs = bid >> 4;          // b*S + s
  int s = bs & (Sdim - 1);
  int b = bs >> 11;
  int j = threadIdx.x;        // 0..191
  size_t qrow = (size_t)bs * 3072 + (size_t)h * 192;
  size_t kvrow = (size_t)bs * 4096 + (size_t)h * 256;
  size_t orow = ((size_t)(b * 16 + h) * Sdim + s) * 192;
  const float LN10000 = 9.210340371976184f;

  float qv;
  if (j < 128) {
    qv = q[qrow + j];
  } else {
    int jj = j - 128;
    int fi = jj & 31;
    float ang = (float)s * __expf(-(2.0f * fi / 64.0f) * LN10000);
    float c = __cosf(ang), sn = __sinf(ang);
    float xv = q[qrow + 128 + jj];
    float other = (jj < 32) ? -q[qrow + 128 + jj + 32] : q[qrow + 128 + jj - 32];
    qv = xv * c + other * sn;
  }
  Qb[orow + j] = f2bf(qv);

  float kvv;
  if (j < 128) {
    kvv = kv[kvrow + j];
  } else {
    int jj = j - 128;
    size_t krow = (size_t)bs * 576 + 512;
    int fi = jj & 31;
    float ang = (float)s * __expf(-(2.0f * fi / 64.0f) * LN10000);
    float c = __cosf(ang), sn = __sinf(ang);
    float xv = kva[krow + jj];
    float other = (jj < 32) ? -kva[krow + jj + 32] : kva[krow + jj - 32];
    kvv = xv * c + other * sn;
  }
  Kb[orow + j] = f2bf(kvv);

  if (j < 128) {
    size_t vrow = ((size_t)(b * 16 + h) * Sdim + s) * 128;
    Vb[vrow + j] = f2bf(kv[kvrow + 128 + j]);
  }
}

// ---------------------------------------------------------------------------
// Causal flash attention. Grid: (S/64, B*H), 256 threads (8 waves).
// Q tile 64 rows, K tile 32 keys, head dim 192, VH 128.
// Q/K tiles staged via GLOBAL_LOAD_ASYNC_TO_LDS_B128 (ASYNCcnt).
// Output written as bf16 [B*S][H*128] (feeds o-proj GEMM directly).
// ---------------------------------------------------------------------------
__global__ __launch_bounds__(256) void flash_attn_kernel(
    const unsigned short* __restrict__ Qb, const unsigned short* __restrict__ Kb,
    const unsigned short* __restrict__ Vb, unsigned short* __restrict__ Out) {
  const int Sdim = 2048;
  int qt = blockIdx.x;
  int bh = blockIdx.y;
  int b = bh >> 4, h = bh & 15;
  int q0 = qt * 64;
  int t = threadIdx.x;
  int w = t >> 5, l = t & 31;

  __shared__ __align__(16) unsigned short Qs[64 * 192];
  __shared__ __align__(16) unsigned short Ks[32 * 192];
  __shared__ __align__(16) unsigned short Vs[128 * 32];   // V transposed [n][k]
  __shared__ __align__(16) float Sp[64 * 32];
  __shared__ __align__(16) unsigned short Pb[64 * 32];
  __shared__ float rowa[64];
  __shared__ float rowl[64];

  {
    unsigned qdst = lds_off(Qs);
    const unsigned short* src = Qb + ((size_t)bh * Sdim + q0) * 192;
#pragma unroll
    for (int i = 0; i < 6; i++) {
      int u = t + 256 * i;                         // 1536 uint4s
      async_copy_b128(qdst + (unsigned)u * 16, (const void*)(src + u * 8));
    }
  }

  float m_i = -1e30f, l_i = 0.0f;    // valid in threads t<64 (row t)
  f32x8 oacc[4];
#pragma unroll
  for (int i = 0; i < 4; i++) oacc[i] = zero8();
  wait_async0();
  __syncthreads();

  int nkt = 2 * qt + 2;              // causal tile bound
  for (int kt = 0; kt < nkt; kt++) {
    int k0 = kt * 32;
    {
      unsigned kdst = lds_off(Ks);
      const unsigned short* src = Kb + ((size_t)bh * Sdim + k0) * 192;
#pragma unroll
      for (int i = 0; i < 3; i++) {
        int u = t + 256 * i;                       // 768 uint4s
        async_copy_b128(kdst + (unsigned)u * 16, (const void*)(src + u * 8));
      }
    }
    // stage V tile transposed: 2 k-rows x 8 n per thread
    {
      int kk = (t >> 4) * 2, nb2 = (t & 15) * 8;
      const unsigned short* v0p = Vb + ((size_t)bh * Sdim + k0 + kk) * 128 + nb2;
      const unsigned short* v1p = v0p + 128;
      unsigned short e0[8], e1[8];
      *(uint4*)(e0) = *(const uint4*)(v0p);
      *(uint4*)(e1) = *(const uint4*)(v1p);
#pragma unroll
      for (int i = 0; i < 8; i++) {
        unsigned pk = (unsigned)e0[i] | ((unsigned)e1[i] << 16);
        *(unsigned*)(Vs + (nb2 + i) * 32 + kk) = pk;
      }
    }
    wait_async0();
    __syncthreads();

    // S = Q K^T * scale, causal mask -> Sp
    {
      int mt = w >> 1, nt = w & 1;
      f32x8 sacc = zero8();
#pragma unroll
      for (int kk = 0; kk < 6; kk++) {
        bf16x16 af = load_frag_a(Qs, 192, mt * 16, kk * 32);
        bf16x16 bf = load_frag_b(Ks, 192, nt * 16, kk * 32);
        sacc = __builtin_amdgcn_wmma_f32_16x16x32_bf16(
            false, af, false, bf, (short)0, sacc, false, false);
      }
      const float sc = 0.07216878364870323f;  // 1/sqrt(192)
      int n_ = l & 15, g = l >> 4;
      int col = k0 + nt * 16 + n_;
#pragma unroll
      for (int r = 0; r < 8; r++) {
        int mrow = mt * 16 + g * 8 + r;
        float v = sacc[r] * sc;
        if (col > q0 + mrow) v = -1e9f;
        Sp[mrow * 32 + nt * 16 + n_] = v;
      }
    }
    __syncthreads();

    // online softmax: thread t<64 owns query row t
    if (t < 64) {
      float mx = m_i;
      for (int jc = 0; jc < 32; jc++) mx = fmaxf(mx, Sp[t * 32 + jc]);
      float alpha = __expf(m_i - mx);
      float ssum = 0.0f;
      for (int jc = 0; jc < 32; jc += 2) {
        float p0 = __expf(Sp[t * 32 + jc] - mx);
        float p1 = __expf(Sp[t * 32 + jc + 1] - mx);
        ssum += p0 + p1;
        *(unsigned*)(Pb + t * 32 + jc) = pack2(p0, p1);
      }
      l_i = l_i * alpha + ssum;
      m_i = mx;
      rowa[t] = alpha;
    }
    __syncthreads();

    // O = alpha*O + P V : wave w owns tiles id=w*4+i (mt=id/8, nt=id%8)
    {
      int g = l >> 4;
#pragma unroll
      for (int i = 0; i < 4; i++) {
        int id = w * 4 + i;
        int mt = id >> 3, nt = id & 7;
#pragma unroll
        for (int r = 0; r < 8; r++) oacc[i][r] *= rowa[mt * 16 + g * 8 + r];
        bf16x16 af = load_frag_a(Pb, 32, mt * 16, 0);
        bf16x16 bf = load_frag_b(Vs, 32, nt * 16, 0);
        oacc[i] = __builtin_amdgcn_wmma_f32_16x16x32_bf16(
            false, af, false, bf, (short)0, oacc[i], false, false);
      }
    }
    __syncthreads();
  }

  if (t < 64) rowl[t] = l_i;
  __syncthreads();
  {
    int n_ = l & 15, g = l >> 4;
#pragma unroll
    for (int i = 0; i < 4; i++) {
      int id = w * 4 + i;
      int mt = id >> 3, nt = id & 7;
#pragma unroll
      for (int r = 0; r < 8; r++) {
        int mrow = mt * 16 + g * 8 + r;
        float v = oacc[i][r] / rowl[mrow];
        size_t srow = (size_t)b * Sdim + q0 + mrow;
        Out[srow * 2048 + h * 128 + nt * 16 + n_] = f2bf(v);
      }
    }
  }
}

// ---------------------------------------------------------------------------
// h = silu(gate) * up -> bf16 (paired stores).
// ---------------------------------------------------------------------------
__global__ __launch_bounds__(256) void silu_mul_bf16_kernel(
    const float* __restrict__ gate, const float* __restrict__ up,
    unsigned short* __restrict__ h, size_t npairs) {
  size_t i = (size_t)blockIdx.x * 256 + threadIdx.x;
  if (i < npairs) {
    float g0 = gate[2 * i], g1 = gate[2 * i + 1];
    float u0 = up[2 * i], u1 = up[2 * i + 1];
    float h0 = (g0 / (1.0f + __expf(-g0))) * u0;
    float h1 = (g1 / (1.0f + __expf(-g1))) * u1;
    *(unsigned*)(h + 2 * i) = pack2(h0, h1);
  }
}

// ---------------------------------------------------------------------------
// Host-side orchestration.
// ---------------------------------------------------------------------------
extern "C" void kernel_launch(void* const* d_in, const int* in_sizes, int n_in,
                              void* d_out, int out_size, void* d_ws, size_t ws_size,
                              hipStream_t stream) {
  (void)in_sizes; (void)n_in; (void)out_size; (void)ws_size;
  const int Btot = 4096;   // B*S rows
  const float* x               = (const float*)d_in[0];
  const float* attn_norm_scale = (const float*)d_in[1];
  const float* q_a_w           = (const float*)d_in[2];
  const float* q_a_norm_scale  = (const float*)d_in[3];
  const float* q_b_w           = (const float*)d_in[4];
  const float* kv_a_w          = (const float*)d_in[5];
  const float* kv_a_norm_scale = (const float*)d_in[6];
  const float* kv_b_w          = (const float*)d_in[7];
  const float* o_w             = (const float*)d_in[8];
  const float* ffn_norm_scale  = (const float*)d_in[9];
  const float* gate_w          = (const float*)d_in[10];
  const float* up_w            = (const float*)d_in[11];
  const float* down_w          = (const float*)d_in[12];
  float* out = (float*)d_out;
  char* ws = (char*)d_ws;

  // --- persistent bf16 transposed weights (126,353,408 B) ---
  unsigned short* q_a_wt  = (unsigned short*)(ws + 0ull);           // [1024][2048]
  unsigned short* q_b_wt  = (unsigned short*)(ws + 4194304ull);     // [3072][1024]
  unsigned short* kv_a_wt = (unsigned short*)(ws + 10485760ull);    // [640][2048] (pad)
  unsigned short* kv_b_wt = (unsigned short*)(ws + 13107200ull);    // [4096][512]
  unsigned short* o_wt    = (unsigned short*)(ws + 17301504ull);    // [2048][2048]
  unsigned short* gate_wt = (unsigned short*)(ws + 25690112ull);    // [8192][2048]
  unsigned short* up_wt   = (unsigned short*)(ws + 59244544ull);    // [8192][2048]
  unsigned short* down_wt = (unsigned short*)(ws + 92798976ull);    // [2048][8192]
  const unsigned long long SB = 126353408ull;                       // scratch base

  // --- phase-1 scratch ---
  unsigned short* xn_bf  = (unsigned short*)(ws + SB + 0ull);          // 4096x2048
  float*          qa     = (float*)(ws + SB + 16777216ull);            // 4096x1024
  unsigned short* qa_bf  = (unsigned short*)(ws + SB + 33554432ull);   // 4096x1024
  float*          qv     = (float*)(ws + SB + 41943040ull);            // 4096x3072
  float*          kva    = (float*)(ws + SB + 92274688ull);            // 4096x576
  unsigned short* ckv_bf = (unsigned short*)(ws + SB + 101711872ull);  // 4096x512
  float*          kvb    = (float*)(ws + SB + 105906176ull);           // 4096x4096
  unsigned short* Qb     = (unsigned short*)(ws + SB + 173015040ull);  // [2][16][2048][192]
  unsigned short* Kb     = (unsigned short*)(ws + SB + 198180864ull);
  unsigned short* Vb     = (unsigned short*)(ws + SB + 223346688ull);  // [2][16][2048][128]
  unsigned short* attn_bf= (unsigned short*)(ws + SB + 240123904ull);  // 4096x2048
  // --- phase-2 scratch (reuses phase-1 region) ---
  unsigned short* xn2_bf = (unsigned short*)(ws + SB + 0ull);          // 4096x2048
  float*          gbuf   = (float*)(ws + SB + 16777216ull);            // 4096x8192
  float*          ubuf   = (float*)(ws + SB + 150994944ull);           // 4096x8192
  unsigned short* h_bf   = (unsigned short*)(ws + SB + 285212672ull);  // 4096x8192

  // --- weight convert + transpose (once per call) ---
  transpose_bf16_kernel<<<dim3(32, 16), 256, 0, stream>>>(q_a_w,  q_a_wt,  2048, 1024, 1024);
  transpose_bf16_kernel<<<dim3(16, 48), 256, 0, stream>>>(q_b_w,  q_b_wt,  1024, 3072, 3072);
  transpose_bf16_kernel<<<dim3(32, 10), 256, 0, stream>>>(kv_a_w, kv_a_wt, 2048, 576, 640);
  transpose_bf16_kernel<<<dim3(8, 64),  256, 0, stream>>>(kv_b_w, kv_b_wt, 512, 4096, 4096);
  transpose_bf16_kernel<<<dim3(32, 32), 256, 0, stream>>>(o_w,    o_wt,    2048, 2048, 2048);
  transpose_bf16_kernel<<<dim3(32, 128),256, 0, stream>>>(gate_w, gate_wt, 2048, 8192, 8192);
  transpose_bf16_kernel<<<dim3(32, 128),256, 0, stream>>>(up_w,   up_wt,   2048, 8192, 8192);
  transpose_bf16_kernel<<<dim3(128, 32),256, 0, stream>>>(down_w, down_wt, 8192, 2048, 2048);

  // --- decoder block ---
  rmsnorm_bf16_kernel<<<Btot, 256, 0, stream>>>(x, xn_bf, attn_norm_scale, 2048, 2048, 2048);
  gemm_bf16_kernel<<<dim3(32, 8), 256, 0, stream>>>(xn_bf, q_a_wt, nullptr, qa, Btot, 1024, 2048, 2048);
  rmsnorm_bf16_kernel<<<Btot, 256, 0, stream>>>(qa, qa_bf, q_a_norm_scale, 1024, 1024, 1024);
  gemm_bf16_kernel<<<dim3(32, 24), 256, 0, stream>>>(qa_bf, q_b_wt, nullptr, qv, Btot, 3072, 1024, 1024);
  gemm_bf16_kernel<<<dim3(32, 5), 256, 0, stream>>>(xn_bf, kv_a_wt, nullptr, kva, Btot, 576, 2048, 2048);
  rmsnorm_bf16_kernel<<<Btot, 256, 0, stream>>>(kva, ckv_bf, kv_a_norm_scale, 512, 576, 512);
  gemm_bf16_kernel<<<dim3(32, 32), 256, 0, stream>>>(ckv_bf, kv_b_wt, nullptr, kvb, Btot, 4096, 512, 512);
  build_qkv_kernel<<<Btot * 16, 192, 0, stream>>>(qv, kvb, kva, Qb, Kb, Vb);
  flash_attn_kernel<<<dim3(32, 32), 256, 0, stream>>>(Qb, Kb, Vb, attn_bf);
  gemm_bf16_kernel<<<dim3(32, 16), 256, 0, stream>>>(attn_bf, o_wt, x, out, Btot, 2048, 2048, 2048);
  rmsnorm_bf16_kernel<<<Btot, 256, 0, stream>>>(out, xn2_bf, ffn_norm_scale, 2048, 2048, 2048);
  gemm_bf16_kernel<<<dim3(32, 64), 256, 0, stream>>>(xn2_bf, gate_wt, nullptr, gbuf, Btot, 8192, 2048, 2048);
  gemm_bf16_kernel<<<dim3(32, 64), 256, 0, stream>>>(xn2_bf, up_wt, nullptr, ubuf, Btot, 8192, 2048, 2048);
  silu_mul_bf16_kernel<<<65536, 256, 0, stream>>>(gbuf, ubuf, h_bf, (size_t)Btot * 4096);
  gemm_bf16_kernel<<<dim3(32, 16), 256, 0, stream>>>(h_bf, down_wt, out, out, Btot, 2048, 8192, 8192);
}
